// DiffPoolLayer_57655640982225
// MI455X (gfx1250) — matrix operand order repaired and checked by reading, not compiled, not run on hardware
//
#include <hip/hip_runtime.h>

typedef __attribute__((ext_vector_type(16))) __bf16 v16bf;
typedef __attribute__((ext_vector_type(8)))  __bf16 v8bf;
typedef __attribute__((ext_vector_type(4)))  __bf16 v4bf;
typedef __attribute__((ext_vector_type(8)))  float  v8f;
typedef __attribute__((ext_vector_type(4)))  float  v4f;

#define NPART 128
#define XS    520   // x_lds row stride (bf16 elems), 16B-aligned rows, bank-spread
#define HSTR  136   // h_t node stride
#define SSTR  136   // s_t node stride

__device__ __forceinline__ v8f vzero8() {
  v8f v;
#pragma unroll
  for (int i = 0; i < 8; i++) v[i] = 0.0f;
  return v;
}

// Build a 16-element bf16 WMMA operand from two 16-byte LDS/global chunks.
__device__ __forceinline__ v16bf ld16(const __bf16* p0, const __bf16* p1) {
  union { v16bf v; v8bf h[2]; } u;
  u.h[0] = *(const v8bf*)p0;
  u.h[1] = *(const v8bf*)p1;
  return u.v;
}

__device__ __forceinline__ v8f wmma_bf16(v16bf a, v16bf b, v8f c) {
  return __builtin_amdgcn_wmma_f32_16x16x32_bf16(false, a, false, b, (short)0, c,
                                                 false, false);
}

// ---------------------------------------------------------------------------
// Kernel 0: convert + transpose weights to bf16 (We_t[n][k], Wp_t[c][k])
// ---------------------------------------------------------------------------
__global__ __launch_bounds__(256) void cvt_weights(
    const float* __restrict__ Wp, const float* __restrict__ We,
    __bf16* __restrict__ Wp_t, __bf16* __restrict__ We_t) {
  const int i = blockIdx.x * 256 + threadIdx.x;
  if (i < 512 * 512) {
    const int n = i >> 9, k = i & 511;
    We_t[i] = (__bf16)We[k * 512 + n];
  } else {
    const int j = i - 512 * 512;   // grid sized exactly: j < 64*512
    const int c = j >> 9, k = j & 511;
    Wp_t[j] = (__bf16)Wp[k * 64 + c];
  }
}

// ---------------------------------------------------------------------------
// Kernel 1: fused  s = softmax(x@Wp+bp), h = x@We+be, partial += s^T h
// ---------------------------------------------------------------------------
__global__ __launch_bounds__(256, 1) void diffpool_main(
    const float* __restrict__ x,
    const __bf16* __restrict__ We_t,
    const __bf16* __restrict__ Wp_t,
    const float* __restrict__ bp,
    const float* __restrict__ be,
    float* __restrict__ partials,
    int N, int nblocks) {
  __shared__ __bf16 x_lds[128 * XS];   // x block, row-major [node][k]
  __shared__ __bf16 h_t[512 * HSTR];   // h block, transposed [f][node]
  __shared__ __bf16 s_t[64 * SSTR];    // s block, transposed [c][node]

  const int tid  = threadIdx.x;
  const int lane = tid & 31;
  const int w    = tid >> 5;      // wave id 0..7
  const int l16  = lane & 15;
  const int half = lane >> 4;

  // persistent pooled accumulators: pool[ct][ft] covers c=16ct.., f=64w+16ft..
  v8f pool[4][4];
#pragma unroll
  for (int a = 0; a < 4; a++)
#pragma unroll
    for (int b = 0; b < 4; b++) pool[a][b] = vzero8();

  for (int blk = blockIdx.x; blk < nblocks; blk += gridDim.x) {
    const int base = blk * 128;

    // ---- stage x block: fp32 global -> bf16 LDS (zero-fill padded rows) ----
#pragma unroll 2
    for (int i = tid; i < 128 * 128; i += 256) {
      const int r  = i >> 7;
      const int c4 = (i & 127) << 2;
      v4f v; v[0] = v[1] = v[2] = v[3] = 0.0f;
      const int node = base + r;
      if (node < N) v = *(const v4f*)(x + (size_t)node * 512 + c4);
      v4bf o;
#pragma unroll
      for (int j = 0; j < 4; j++) o[j] = (__bf16)v[j];
      *(v4bf*)&x_lds[r * XS + c4] = o;
    }
    __syncthreads();

    // ---- s pass: wave w owns rows 16w..16w+15, all 64 clusters ----
    {
      v8f acc[4];
#pragma unroll
      for (int ct = 0; ct < 4; ct++) acc[ct] = vzero8();
      const __bf16* xrow = &x_lds[(16 * w + l16) * XS];
#pragma unroll 1
      for (int k0 = 0; k0 < 512; k0 += 32) {
        const __bf16* ap = xrow + k0 + 16 * half;
        v16bf A = ld16(ap, ap + 8);
#pragma unroll
        for (int ct = 0; ct < 4; ct++) {
          const __bf16* bpt = Wp_t + (size_t)(16 * ct + l16) * 512 + k0 + 16 * half;
          v16bf B = ld16(bpt, bpt + 8);
          acc[ct] = wmma_bf16(A, B, acc[ct]);
        }
      }
#pragma unroll
      for (int ct = 0; ct < 4; ct++) {
        const float bb = bp[16 * ct + l16];
#pragma unroll
        for (int r = 0; r < 8; r++) acc[ct][r] += bb;
      }
      // softmax over the 64 clusters of each row (4 regs x 16 lanes per row)
      float rs[8];
#pragma unroll
      for (int r = 0; r < 8; r++) {
        float m = fmaxf(fmaxf(acc[0][r], acc[1][r]), fmaxf(acc[2][r], acc[3][r]));
        m = fmaxf(m, __shfl_xor(m, 1, 32));
        m = fmaxf(m, __shfl_xor(m, 2, 32));
        m = fmaxf(m, __shfl_xor(m, 4, 32));
        m = fmaxf(m, __shfl_xor(m, 8, 32));
        float e0 = __expf(acc[0][r] - m);
        float e1 = __expf(acc[1][r] - m);
        float e2 = __expf(acc[2][r] - m);
        float e3 = __expf(acc[3][r] - m);
        acc[0][r] = e0; acc[1][r] = e1; acc[2][r] = e2; acc[3][r] = e3;
        float s = (e0 + e1) + (e2 + e3);
        s += __shfl_xor(s, 1, 32);
        s += __shfl_xor(s, 2, 32);
        s += __shfl_xor(s, 4, 32);
        s += __shfl_xor(s, 8, 32);
        rs[r] = 1.0f / s;
      }
      // store s transposed (bf16); zero out padded rows so they drop from s^T h
#pragma unroll
      for (int ct = 0; ct < 4; ct++) {
#pragma unroll
        for (int r = 0; r < 8; r++) {
          const int row = 16 * w + r + 8 * half;
          const float sv = acc[ct][r] * rs[r];
          s_t[(16 * ct + l16) * SSTR + row] =
              (base + row < N) ? (__bf16)sv : (__bf16)0.0f;
        }
      }
    }

    // ---- h pass: wave w owns feature columns [64w, 64w+64) ----
#pragma unroll 1
    for (int nt = 0; nt < 4; nt++) {
      const int col = 64 * w + 16 * nt + l16;
      v8f acc[8];
#pragma unroll
      for (int mt = 0; mt < 8; mt++) acc[mt] = vzero8();
      const __bf16* bcol = We_t + (size_t)col * 512;
#pragma unroll 1
      for (int k0 = 0; k0 < 512; k0 += 32) {
        const __bf16* bpt = bcol + k0 + 16 * half;
        v16bf B = ld16(bpt, bpt + 8);            // B held across 8 M-tiles
#pragma unroll
        for (int mt = 0; mt < 8; mt++) {
          const __bf16* ap = &x_lds[(16 * mt + l16) * XS + k0 + 16 * half];
          v16bf A = ld16(ap, ap + 8);
          acc[mt] = wmma_bf16(A, B, acc[mt]);
        }
      }
      const float bv = be[col];
#pragma unroll
      for (int mt = 0; mt < 8; mt++)
#pragma unroll
        for (int r = 0; r < 8; r++)
          h_t[col * HSTR + 16 * mt + r + 8 * half] = (__bf16)(acc[mt][r] + bv);
    }
    __syncthreads();

    // ---- pooled accumulate: pool += s^T(64x128) @ h(128x512) ----
#pragma unroll 1
    for (int k0 = 0; k0 < 128; k0 += 32) {
      v16bf Bt[4];
#pragma unroll
      for (int ft = 0; ft < 4; ft++) {
        const __bf16* bpt = &h_t[(64 * w + 16 * ft + l16) * HSTR + k0 + 16 * half];
        Bt[ft] = ld16(bpt, bpt + 8);
      }
#pragma unroll
      for (int ct = 0; ct < 4; ct++) {
        const __bf16* ap = &s_t[(16 * ct + l16) * SSTR + k0 + 16 * half];
        v16bf A = ld16(ap, ap + 8);
#pragma unroll
        for (int ft = 0; ft < 4; ft++)
          pool[ct][ft] = wmma_bf16(A, Bt[ft], pool[ct][ft]);
      }
    }
    __syncthreads();   // protect s_t/h_t before next block rewrites them
  }

  // ---- write this workgroup's partial pooled [64][512] ----
  float* dst = partials + (size_t)blockIdx.x * (64 * 512);
#pragma unroll
  for (int ct = 0; ct < 4; ct++)
#pragma unroll
    for (int ft = 0; ft < 4; ft++)
#pragma unroll
      for (int r = 0; r < 8; r++) {
        const int c = 16 * ct + r + 8 * half;
        const int f = 64 * w + 16 * ft + l16;
        dst[c * 512 + f] = pool[ct][ft][r];
      }
}

// ---------------------------------------------------------------------------
// Kernel 2: deterministic reduction of partials -> pooled [64*512]
// ---------------------------------------------------------------------------
__global__ __launch_bounds__(256) void reduce_partials(
    const float* __restrict__ partials, float* __restrict__ pooled) {
  const int i = blockIdx.x * 256 + threadIdx.x;   // 32768 total
  float s = 0.0f;
#pragma unroll 4
  for (int p = 0; p < NPART; p++) s += partials[(size_t)p * 32768 + i];
  pooled[i] = s;
}

// ---------------------------------------------------------------------------
// Kernel 3: out[c][o] = pooled[c] . Wo[:,o] + bo[o]     (64 x 256, fp32)
// ---------------------------------------------------------------------------
__global__ __launch_bounds__(256) void out_proj(
    const float* __restrict__ pooled, const float* __restrict__ Wo,
    const float* __restrict__ bo, float* __restrict__ out) {
  const int c = blockIdx.x;
  const int o = threadIdx.x;
  float acc = bo[o];
#pragma unroll 4
  for (int f = 0; f < 512; f++)
    acc = fmaf(pooled[c * 512 + f], Wo[f * 256 + o], acc);
  out[c * 256 + o] = acc;
}

extern "C" void kernel_launch(void* const* d_in, const int* in_sizes, int n_in,
                              void* d_out, int out_size, void* d_ws, size_t ws_size,
                              hipStream_t stream) {
  const float* x  = (const float*)d_in[0];
  // d_in[1] = edge_index, d_in[2] = batch : dead inputs (adj outputs discarded)
  const float* Wp = (const float*)d_in[3];
  const float* bp = (const float*)d_in[4];
  const float* We = (const float*)d_in[5];
  const float* be = (const float*)d_in[6];
  const float* Wo = (const float*)d_in[7];
  const float* bo = (const float*)d_in[8];

  const int N = in_sizes[0] / 512;
  const int nblocks = (N + 127) / 128;

  char* ws = (char*)d_ws;
  __bf16* We_t    = (__bf16*)ws;                                   // 512 KiB
  __bf16* Wp_t    = (__bf16*)(ws + 524288);                        //  64 KiB
  float*  partials = (float*)(ws + 524288 + 65536);                // 16 MiB
  float*  pooled   = (float*)(ws + 524288 + 65536 +
                              (size_t)NPART * 64 * 512 * 4);       // 128 KiB

  cvt_weights<<<(512 * 512 + 64 * 512) / 256, 256, 0, stream>>>(Wp, We, Wp_t, We_t);
  diffpool_main<<<NPART, 256, 0, stream>>>(x, We_t, Wp_t, bp, be, partials,
                                           N, nblocks);
  reduce_partials<<<128, 256, 0, stream>>>(partials, pooled);
  out_proj<<<64, 256, 0, stream>>>(pooled, Wo, bo, (float*)d_out);
}
